// MyLSTM_10075993277120
// MI455X (gfx1250) — matrix operand order repaired and checked by reading, not compiled, not run on hardware
//
#include <hip/hip_runtime.h>

typedef __attribute__((ext_vector_type(16))) __bf16 v16bf;
typedef __attribute__((ext_vector_type(4)))  __bf16 v4bf;
typedef __attribute__((ext_vector_type(8)))  float  v8f;

#define B_DIM  4096
#define N_IN   2048
#define N_OUTD 2048
#define K_DIM  4096   // N_IN + N_OUT
#define MT     128    // batch rows per block
#define NT     32     // hidden cols per block (x4 gates -> 128 gate cols)
#define KT     64     // K per staging round (2 WMMA k-steps)
#define LDS_STRIDE 72 // bf16 elems per row (144B): conflict-free b128 reads
#define NROUNDS (K_DIM / KT)   // 64

union SMem {
    struct { __bf16 a[MT * LDS_STRIDE]; __bf16 b[MT * LDS_STRIDE]; } st; // 36 KB
    float g[MT * 128];                                                   // 64 KB
};

__global__ __launch_bounds__(256)
void lstm_fused_wmma_bf16(const float* __restrict__ c_old,
                          const float* __restrict__ h_old,
                          const float* __restrict__ x,
                          const float* __restrict__ Wf, const float* __restrict__ bfv_,
                          const float* __restrict__ Wi, const float* __restrict__ biv_,
                          const float* __restrict__ Wo, const float* __restrict__ bov_,
                          const float* __restrict__ Wg, const float* __restrict__ bgv_,
                          float* __restrict__ out_c, float* __restrict__ out_h)
{
    __shared__ SMem sm;

    const int t    = threadIdx.x;
    const int lane = t & 31;
    const int wave = t >> 5;        // 0..7
    const int wr   = wave >> 2;     // 0..1 : row half of 128
    const int wc   = wave & 3;      // 0..3 : gate index (f,i,o,g)
    const int half = lane >> 4;     // 0..1
    const int m    = lane & 15;

    const int mbase = blockIdx.y * MT;   // batch row base
    const int nbase = blockIdx.x * NT;   // hidden col base

    // ---- hoisted per-pass staging offsets (pass p covers tile rows 16p..16p+15;
    //      gate = p/2 is a compile-time constant after unroll) ----
    const int rsub = t >> 4;          // 0..15 row within pass
    const int c4   = (t & 15) << 2;   // 0,4,...,60
    int aoff[8], woff[8];
    #pragma unroll
    for (int p = 0; p < 8; ++p) {
        const int r = p * 16 + rsub;                       // 0..127
        aoff[p] = (mbase + r) * N_IN + c4;                 // x/h both 2048 wide
        woff[p] = (nbase + (r & 31)) * K_DIM + c4;
    }
    // one-instruction whole-tile prefetch addresses (thread t covers row t/2, half t&1)
    const int pr     = t >> 1;
    const int pcol   = (t & 1) << 5;                       // 0 or 32 floats (128B)
    const long paoff = (long)(mbase + pr) * N_IN + pcol;
    const float* pwbase = (pr < 32) ? Wf : (pr < 64) ? Wi : (pr < 96) ? Wo : Wg;
    const float* pwptr  = pwbase + (long)(nbase + (pr & 31)) * K_DIM + pcol;

    v8f acc[4][2] = {};  // 4 M-tiles x 2 N-tiles of 16x16 f32

    for (int r2 = 0; r2 < NROUNDS; ++r2) {
        const float* abase = (r2 < NROUNDS / 2) ? x : h_old;   // uniform select
        const int k0a = (r2 & (NROUNDS / 2 - 1)) * KT;         // col within x or h
        const int k0w = r2 * KT;                               // col within W

        __syncthreads();   // previous round's LDS reads complete
        // ---- stage A and B tiles, f32 -> bf16 ----
        #pragma unroll
        for (int p = 0; p < 8; ++p) {
            const float4 fa = *(const float4*)(abase + aoff[p] + k0a);
            const float* wb = (p < 2) ? Wf : (p < 4) ? Wi : (p < 6) ? Wo : Wg;
            const float4 fw = *(const float4*)(wb + woff[p] + k0w);
            const int r = p * 16 + rsub;
            v4bf va; va.x = (__bf16)fa.x; va.y = (__bf16)fa.y;
                     va.z = (__bf16)fa.z; va.w = (__bf16)fa.w;
            v4bf vb; vb.x = (__bf16)fw.x; vb.y = (__bf16)fw.y;
                     vb.z = (__bf16)fw.z; vb.w = (__bf16)fw.w;
            *(v4bf*)&sm.st.a[r * LDS_STRIDE + c4] = va;
            *(v4bf*)&sm.st.b[r * LDS_STRIDE + c4] = vb;
        }
        // ---- prefetch next round's tiles into WGP-level cache ----
        {
            const int rn  = (r2 + 1 < NROUNDS) ? r2 + 1 : NROUNDS - 1;
            const float* abn = (rn < NROUNDS / 2) ? x : h_old;
            __builtin_prefetch(abn + paoff + (rn & (NROUNDS / 2 - 1)) * KT, 0, 3);
            __builtin_prefetch(pwptr + rn * KT, 0, 3);
        }
        __syncthreads();

        // ---- fragments + WMMA: batch ALL 12 ds loads, then 8 back-to-back WMMAs ----
        #pragma unroll
        for (int ks = 0; ks < 2; ++ks) {
            v16bf b0, b1, af[4];
            {
                const __bf16* pb = &sm.st.b[(wc * 32 + m) * LDS_STRIDE]
                                   + ks * 32 + 16 * half;            // K=16h..16h+15
                ((uint4*)&b0)[0] = *(const uint4*)(pb);
                ((uint4*)&b0)[1] = *(const uint4*)(pb + 8);
            }
            {
                const __bf16* pb = &sm.st.b[(wc * 32 + 16 + m) * LDS_STRIDE]
                                   + ks * 32 + 16 * half;
                ((uint4*)&b1)[0] = *(const uint4*)(pb);
                ((uint4*)&b1)[1] = *(const uint4*)(pb + 8);
            }
            #pragma unroll
            for (int i = 0; i < 4; ++i) {
                const __bf16* pa = &sm.st.a[(wr * 64 + i * 16 + m) * LDS_STRIDE]
                                   + ks * 32;
                ((uint4*)&af[i])[0] = *(const uint4*)(pa + 16 * half);      // K=16h..
                ((uint4*)&af[i])[1] = *(const uint4*)(pa + 16 * half + 8);  // K=16h+8..
            }
            #pragma unroll
            for (int i = 0; i < 4; ++i) {
                acc[i][0] = __builtin_amdgcn_wmma_f32_16x16x32_bf16(
                    false, af[i], false, b0, (short)0, acc[i][0], false, false);
                acc[i][1] = __builtin_amdgcn_wmma_f32_16x16x32_bf16(
                    false, af[i], false, b1, (short)0, acc[i][1], false, false);
            }
        }
    }

    // ---- spill gate accumulators to LDS (C/D layout: VGPR v -> M = v + 8*half) ----
    __syncthreads();
    #pragma unroll
    for (int i = 0; i < 4; ++i)
        #pragma unroll
        for (int j = 0; j < 2; ++j)
            #pragma unroll
            for (int v = 0; v < 8; ++v)
                sm.g[(wr * 64 + i * 16 + half * 8 + v) * 128
                     + wc * 32 + j * 16 + m] = acc[i][j][v];
    __syncthreads();

    // ---- fused LSTM pointwise: bias + gates + state update ----
    const int jn = t & 31;
    const int rb = (t >> 5) * 16;
    const int n  = nbase + jn;
    const float bf_ = bfv_[n], bi_ = biv_[n], bo_ = bov_[n], bg_ = bgv_[n];

    #pragma unroll 4
    for (int rr = 0; rr < 16; ++rr) {
        const int r = rb + rr;
        const float* gr = &sm.g[r * 128];
        const float zf = gr[jn]      + bf_;
        const float zi = gr[32 + jn] + bi_;
        const float zo = gr[64 + jn] + bo_;
        const float zg = gr[96 + jn] + bg_;

        const float fg = 1.0f / (1.0f + __expf(-zf));
        const float ig = 1.0f / (1.0f + __expf(-zi));
        const float og = 1.0f / (1.0f + __expf(-zo));
        const float e2g = __expf(2.0f * zg);
        const float gg  = (e2g - 1.0f) / (e2g + 1.0f);

        const size_t off = (size_t)(mbase + r) * N_OUTD + n;
        const float c = fg * c_old[off] + ig * gg;
        const float e2c = __expf(2.0f * c);
        const float th  = (e2c - 1.0f) / (e2c + 1.0f);

        out_c[off] = c;
        out_h[off] = og * th;
    }
}

extern "C" void kernel_launch(void* const* d_in, const int* in_sizes, int n_in,
                              void* d_out, int out_size, void* d_ws, size_t ws_size,
                              hipStream_t stream) {
    (void)in_sizes; (void)n_in; (void)out_size; (void)d_ws; (void)ws_size;
    const float* c_old = (const float*)d_in[0];
    const float* h_old = (const float*)d_in[1];
    const float* x     = (const float*)d_in[2];
    const float* Wf    = (const float*)d_in[3];
    const float* bf    = (const float*)d_in[4];
    const float* Wi    = (const float*)d_in[5];
    const float* bi    = (const float*)d_in[6];
    const float* Wo    = (const float*)d_in[7];
    const float* bo    = (const float*)d_in[8];
    const float* Wg    = (const float*)d_in[9];
    const float* bg    = (const float*)d_in[10];
    // d_in[11] = mode (unused; decode path is identical math)

    float* out_c = (float*)d_out;
    float* out_h = out_c + (size_t)B_DIM * N_OUTD;

    dim3 grid(N_OUTD / NT, B_DIM / MT);   // 64 x 32
    dim3 block(256);
    lstm_fused_wmma_bf16<<<grid, block, 0, stream>>>(
        c_old, h_old, x, Wf, bf, Wi, bi, Wo, bo, Wg, bg, out_c, out_h);
}
